// TransformerModel_39874476376281
// MI455X (gfx1250) — matrix operand order repaired and checked by reading, not compiled, not run on hardware
//
#include <hip/hip_runtime.h>
#include <type_traits>

// ---------------------------------------------------------------------------
// Transformer forward for MI455X (gfx1250, wave32, WMMA).
// All GEMMs: bf16 x bf16 -> fp32 via v_wmma_f32_16x16x32_bf16, operands staged
// through LDS with double-buffered GLOBAL_LOAD_ASYNC_TO_LDS_B128 (ASYNCcnt).
// Softmax / LayerNorm / residual kept in fp32.
// ---------------------------------------------------------------------------

typedef __attribute__((ext_vector_type(8)))  __bf16 bf16x8;
typedef __attribute__((ext_vector_type(16))) __bf16 bf16x16;
typedef __attribute__((ext_vector_type(8)))  float  f32x8;
typedef __attribute__((ext_vector_type(4)))  int    v4i;

// Probe-revealed builtin parameter types: global side is a v4i addrspace(1)*.
typedef __attribute__((address_space(1))) v4i* g4_t;   // global 16B chunk
typedef __attribute__((address_space(3))) v4i* l4_t;   // LDS 16B chunk

#if __has_builtin(__builtin_amdgcn_global_load_async_to_lds_b128) && \
    __has_builtin(__builtin_amdgcn_s_wait_asynccnt)
#define ASYNC_LDS 1
#else
#define ASYNC_LDS 0
#endif

constexpr int   V_   = 32000;
constexpr int   D_   = 1024;
constexpr int   L_   = 8;
constexpr int   B_   = 2;
constexpr int   T_   = 1024;
constexpr int   NTOK = B_ * T_;           // 2048 token rows

// GEMM tiling: 256 threads = 8 waves; block tile 128x128; wave tile 64x32
constexpr int BM = 128, BN = 128, BK = 32;
constexpr int LPAD = 40;   // LDS row pitch in bf16 (80B): conflict-free b128 reads

__device__ __forceinline__ bf16x16 frag_cat(bf16x8 lo, bf16x8 hi) {
  return __builtin_shufflevector(lo, hi, 0,1,2,3,4,5,6,7,8,9,10,11,12,13,14,15);
}

__device__ __forceinline__ l4_t to_lds(const void* p) {
  // Generic LDS address truncates to the 32-bit LDS offset (ISA 10.2).
  return (l4_t)(uint32_t)(uintptr_t)p;
}
__device__ __forceinline__ g4_t to_glob(const void* p) {
  return (g4_t)(uintptr_t)p;
}

// ---------------------------------------------------------------------------
// Generic WMMA GEMM:  C[M,N] = A[M,K] * BT[N,K]^T  (+ bias)
//   A  : bf16 row-major, leading dim lda   (K contiguous)
//   BT : bf16 row-major, leading dim ldb   (K contiguous; i.e. B column-major)
//   BIAS: 0=none, 1=per-column (bias[n]), 2=per-row (bias[m])
//   OUTBF16: emit bf16 (activations feeding the next GEMM) else fp32
//   CAUSAL: skip blocks entirely above the diagonal (for Q*K^T scores)
// M,N implied by grid (multiples of 128); K runtime (multiple of 32).
// ---------------------------------------------------------------------------
template <int BIAS, bool OUTBF16, bool CAUSAL>
__global__ __launch_bounds__(256) void gemm_bt(
    const __bf16* __restrict__ A,  long long strideA, int lda,
    const __bf16* __restrict__ Bt, long long strideB, int ldb,
    void* __restrict__ Cv,         long long strideC, int ldc,
    const float* __restrict__ bias, int K)
{
  const int m0 = blockIdx.y * BM;
  const int n0 = blockIdx.x * BN;
  if (CAUSAL && n0 >= m0 + BM) return;   // fully masked tile (uniform per block)

  const int bz = blockIdx.z;
  A  += (size_t)bz * strideA;
  Bt += (size_t)bz * strideB;

  const int tid  = threadIdx.x;
  const int wave = tid >> 5;
  const int lane = tid & 31;
  const int wmL = (wave & 1) * 64;       // wave tile origin within block tile
  const int wnL = (wave >> 1) * 32;
  const int fr  = lane & 15;

  // Double-buffered LDS slabs: 128 rows x 32 k, rows padded to LPAD bf16.
  __shared__ __align__(16) __bf16 sA[2][BM * LPAD];
  __shared__ __align__(16) __bf16 sB[2][BN * LPAD];

  // Stage one 128x32 slab of A and of B: 512 16B chunks each; 256 threads x2.
  auto stage = [&](int kk, int buf) {
    #pragma unroll
    for (int p = 0; p < 2; ++p) {
      const int c   = tid + p * 256;
      const int row = c >> 2;            // 0..127
      const int kc  = (c & 3) * 8;       // bf16 offset within slab: 0,8,16,24
      const __bf16* ga = A  + (size_t)(m0 + row) * lda + kk + kc;
      const __bf16* gb = Bt + (size_t)(n0 + row) * ldb + kk + kc;
      __bf16* la = &sA[buf][row * LPAD + kc];
      __bf16* lb = &sB[buf][row * LPAD + kc];
#if ASYNC_LDS
      __builtin_amdgcn_global_load_async_to_lds_b128(to_glob(ga), to_lds(la), 0, 0);
      __builtin_amdgcn_global_load_async_to_lds_b128(to_glob(gb), to_lds(lb), 0, 0);
#else
      *(int4*)la = *(const int4*)ga;     // global_load_b128 + ds_store_b128
      *(int4*)lb = *(const int4*)gb;
#endif
    }
  };

  f32x8 acc[4][2];
  #pragma unroll
  for (int i = 0; i < 4; ++i)
    #pragma unroll
    for (int j = 0; j < 2; ++j)
      acc[i][j] = (f32x8){0.f,0.f,0.f,0.f,0.f,0.f,0.f,0.f};

  const int nk = K >> 5;
  stage(0, 0);

  for (int kt = 0; kt < nk; ++kt) {
    const int buf = kt & 1;
#if ASYNC_LDS
    __builtin_amdgcn_s_wait_asynccnt(0);   // this wave's slab writes landed
#endif
    __syncthreads();                        // everyone's slab visible; prev compute done
    if (kt + 1 < nk) stage((kt + 1) << 5, buf ^ 1);  // DMA next slab during compute

    bf16x16 af[4];
    bf16x16 bfr[2];
    // A-fragment (16-bit A 16x32): lanes 0-15 -> K {0..7,16..23},
    // lanes 16-31 -> K {8..15,24..31}, row = lane&15.
    #pragma unroll
    for (int i = 0; i < 4; ++i) {
      const __bf16* p = &sA[buf][(wmL + 16 * i + fr) * LPAD + ((lane & 16) ? 8 : 0)];
      af[i] = frag_cat(*(const bf16x8*)p, *(const bf16x8*)(p + 16));
    }
    // B-fragment (16-bit B 32x16): lanes 0-15 -> K {0..7,16..23}, lanes 16-31 -> K {8..15,24..31}.
    #pragma unroll
    for (int j = 0; j < 2; ++j) {
      const __bf16* p = &sB[buf][(wnL + 16 * j + fr) * LPAD + ((lane & 16) ? 8 : 0)];
      bfr[j] = frag_cat(*(const bf16x8*)p, *(const bf16x8*)(p + 16));
    }
    #pragma unroll
    for (int i = 0; i < 4; ++i)
      #pragma unroll
      for (int j = 0; j < 2; ++j)
        acc[i][j] = __builtin_amdgcn_wmma_f32_16x16x32_bf16(
            false, af[i], false, bfr[j], (short)0, acc[i][j], false, false);
  }

  // C/D layout: VGPR r, lanes 0-15 -> (M=r, N=lane), lanes 16-31 -> (M=8+r)
  using CT = typename std::conditional<OUTBF16, __bf16, float>::type;
  CT* C = (CT*)Cv + (size_t)bz * strideC;
  const int cl    = lane & 15;
  const int rbase = (lane & 16) ? 8 : 0;
  #pragma unroll
  for (int j = 0; j < 2; ++j) {
    const int col = wnL + n0 + 16 * j + cl;
    const float bn = (BIAS == 1) ? bias[col] : 0.f;
    #pragma unroll
    for (int i = 0; i < 4; ++i) {
      const int row0 = wmL + m0 + 16 * i + rbase;
      #pragma unroll
      for (int r = 0; r < 8; ++r) {
        const int row = row0 + r;
        float v = acc[i][j][r] + bn;
        if (BIAS == 2) v += bias[row];
        C[(size_t)row * ldc + col] = (CT)v;
      }
    }
  }
}

// ---------------------------------------------------------------------------
// Elementwise / reduction kernels
// ---------------------------------------------------------------------------
__global__ void f32_to_bf16(const float* __restrict__ in,
                            __bf16* __restrict__ out, int n) {
  for (int i = blockIdx.x * blockDim.x + threadIdx.x; i < n;
       i += gridDim.x * blockDim.x)
    out[i] = (__bf16)in[i];
}

// x[b,t,d] = embed[src[b,t],d] + pe[b,0,d]   (faithful to the reference's
// pe[:B] batch-indexed positional encoding). Writes fp32 + bf16 copies.
__global__ void embed_pe(const int* __restrict__ src,
                         const float* __restrict__ pe,
                         const float* __restrict__ emb,
                         float* __restrict__ x, __bf16* __restrict__ xb) {
  const int n = NTOK * D_;
  for (int i = blockIdx.x * blockDim.x + threadIdx.x; i < n;
       i += gridDim.x * blockDim.x) {
    const int d  = i & (D_ - 1);
    const int bt = i >> 10;            // token row (b*T + t)
    const int b  = bt >> 10;           // batch
    const float v = emb[(size_t)src[bt] * D_ + d] + pe[b * D_ + d];
    x[i]  = v;
    xb[i] = (__bf16)v;
  }
}

// Row softmax with causal mask: masked lanes contribute exact bf16 zeros so
// the downstream o = P*V GEMM needs no masking.
__global__ __launch_bounds__(256) void softmax_causal(
    const float* __restrict__ scores, __bf16* __restrict__ probs) {
  const int t = blockIdx.x, b = blockIdx.y;
  const float* srow = scores + ((size_t)b * T_ + t) * T_;
  __bf16* prow = probs + ((size_t)b * T_ + t) * T_;
  const int tid = threadIdx.x, wave = tid >> 5, lane = tid & 31;
  __shared__ float redm[8], reds[8];

  float v[4];
  float mx = -3.0e38f;
  #pragma unroll
  for (int i = 0; i < 4; ++i) {
    const int s = tid + i * 256;
    v[i] = (s <= t) ? srow[s] : -3.0e38f;
    mx = fmaxf(mx, v[i]);
  }
  #pragma unroll
  for (int o = 16; o > 0; o >>= 1) mx = fmaxf(mx, __shfl_xor(mx, o, 32));
  if (lane == 0) redm[wave] = mx;
  __syncthreads();
  mx = redm[0];
  #pragma unroll
  for (int i = 1; i < 8; ++i) mx = fmaxf(mx, redm[i]);

  float e[4];
  float sum = 0.f;
  #pragma unroll
  for (int i = 0; i < 4; ++i) {
    const int s = tid + i * 256;
    e[i] = (s <= t) ? __expf(v[i] - mx) : 0.f;
    sum += e[i];
  }
  #pragma unroll
  for (int o = 16; o > 0; o >>= 1) sum += __shfl_xor(sum, o, 32);
  if (lane == 0) reds[wave] = sum;
  __syncthreads();
  sum = 0.f;
  #pragma unroll
  for (int i = 0; i < 8; ++i) sum += reds[i];
  const float inv = 1.0f / sum;
  #pragma unroll
  for (int i = 0; i < 4; ++i) prow[tid + i * 256] = (__bf16)(e[i] * inv);
}

// x += LN_custom(o) * lnw + lnb   (eps=1e-6 inside sqrt, biased var).
// Updates fp32 x in place and refreshes its bf16 mirror.
__global__ __launch_bounds__(256) void ln_residual(
    float* __restrict__ x, __bf16* __restrict__ xb,
    const float* __restrict__ o,
    const float* __restrict__ lnw, const float* __restrict__ lnb) {
  const int row = blockIdx.x;
  const float* orow = o + (size_t)row * D_;
  const int tid = threadIdx.x, wave = tid >> 5, lane = tid & 31;
  __shared__ float r1[8], r2[8];

  float val[4];
  float s = 0.f, ss = 0.f;
  #pragma unroll
  for (int i = 0; i < 4; ++i) {
    val[i] = orow[tid + i * 256];
    s += val[i];
    ss += val[i] * val[i];
  }
  #pragma unroll
  for (int o2 = 16; o2 > 0; o2 >>= 1) {
    s  += __shfl_xor(s,  o2, 32);
    ss += __shfl_xor(ss, o2, 32);
  }
  if (lane == 0) { r1[wave] = s; r2[wave] = ss; }
  __syncthreads();
  s = 0.f; ss = 0.f;
  #pragma unroll
  for (int i = 0; i < 8; ++i) { s += r1[i]; ss += r2[i]; }
  const float mu  = s  * (1.0f / D_);
  const float var = ss * (1.0f / D_) - mu * mu;
  const float inv = 1.0f / sqrtf(var + 1e-6f);
  #pragma unroll
  for (int i = 0; i < 4; ++i) {
    const int d = tid + i * 256;
    const size_t idx = (size_t)row * D_ + d;
    const float xv = x[idx] + (val[i] - mu) * inv * lnw[d] + lnb[d];
    x[idx]  = xv;
    xb[idx] = (__bf16)xv;
  }
}

// Final nn.LayerNorm (eps=1e-5) -> bf16 activations for the logits GEMM.
__global__ __launch_bounds__(256) void final_ln(
    const float* __restrict__ x, __bf16* __restrict__ xnb,
    const float* __restrict__ fw, const float* __restrict__ fb) {
  const int row = blockIdx.x;
  const float* xrow = x + (size_t)row * D_;
  const int tid = threadIdx.x, wave = tid >> 5, lane = tid & 31;
  __shared__ float r1[8], r2[8];

  float val[4];
  float s = 0.f, ss = 0.f;
  #pragma unroll
  for (int i = 0; i < 4; ++i) {
    val[i] = xrow[tid + i * 256];
    s += val[i];
    ss += val[i] * val[i];
  }
  #pragma unroll
  for (int o2 = 16; o2 > 0; o2 >>= 1) {
    s  += __shfl_xor(s,  o2, 32);
    ss += __shfl_xor(ss, o2, 32);
  }
  if (lane == 0) { r1[wave] = s; r2[wave] = ss; }
  __syncthreads();
  s = 0.f; ss = 0.f;
  #pragma unroll
  for (int i = 0; i < 8; ++i) { s += r1[i]; ss += r2[i]; }
  const float mu  = s  * (1.0f / D_);
  const float var = ss * (1.0f / D_) - mu * mu;
  const float inv = rsqrtf(var + 1e-5f);
  #pragma unroll
  for (int i = 0; i < 4; ++i) {
    const int d = tid + i * 256;
    xnb[(size_t)row * D_ + d] = (__bf16)((val[i] - mu) * inv * fw[d] + fb[d]);
  }
}

// ---------------------------------------------------------------------------
// Host launcher
// ---------------------------------------------------------------------------
extern "C" void kernel_launch(void* const* d_in, const int* in_sizes, int n_in,
                              void* d_out, int out_size, void* d_ws, size_t ws_size,
                              hipStream_t stream) {
  (void)in_sizes; (void)n_in; (void)out_size; (void)ws_size;

  const int*   src  = (const int*)  d_in[0];
  const float* pe   = (const float*)d_in[1];
  const float* emb  = (const float*)d_in[2];
  const float* qw   = (const float*)d_in[3];
  const float* qb   = (const float*)d_in[4];
  const float* kw   = (const float*)d_in[5];
  const float* kb   = (const float*)d_in[6];
  const float* vw   = (const float*)d_in[7];
  const float* vb   = (const float*)d_in[8];
  const float* lnw  = (const float*)d_in[9];
  const float* lnb  = (const float*)d_in[10];
  const float* fw   = (const float*)d_in[11];
  const float* fb   = (const float*)d_in[12];
  const float* dw   = (const float*)d_in[13];
  const float* db   = (const float*)d_in[14];
  float* out = (float*)d_out;

  // Workspace carve-up (all offsets 256B aligned by construction).
  char* w = (char*)d_ws;
  size_t off = 0;
  auto carve = [&](size_t bytes) { char* p = w + off; off += (bytes + 255) & ~(size_t)255; return p; };
  float*  xf     = (float*) carve((size_t)NTOK * D_ * 4);   // residual stream fp32
  __bf16* xb16   = (__bf16*)carve((size_t)NTOK * D_ * 2);   // bf16 mirror of x
  __bf16* xnb16  = (__bf16*)carve((size_t)NTOK * D_ * 2);   // final-LN bf16
  __bf16* q16    = (__bf16*)carve((size_t)NTOK * D_ * 2);
  __bf16* k16    = (__bf16*)carve((size_t)NTOK * D_ * 2);
  __bf16* vt16   = (__bf16*)carve((size_t)B_ * D_ * T_ * 2); // V^T per batch [D,T]
  float*  sc     = (float*) carve((size_t)B_ * T_ * T_ * 4); // attention scores
  __bf16* pr16   = (__bf16*)carve((size_t)B_ * T_ * T_ * 2); // softmax probs
  float*  of     = (float*) carve((size_t)NTOK * D_ * 4);    // attention output
  __bf16* wq16   = (__bf16*)carve((size_t)D_ * D_ * 2);      // per-layer weights
  __bf16* wk16   = (__bf16*)carve((size_t)D_ * D_ * 2);
  __bf16* wv16   = (__bf16*)carve((size_t)D_ * D_ * 2);
  __bf16* dw16   = (__bf16*)carve((size_t)V_ * D_ * 2);      // decoder weight bf16

  const long long TD = (long long)T_ * D_;
  const long long TT = (long long)T_ * T_;
  const long long DT = (long long)D_ * T_;

  // Decoder weight -> bf16 (once per call), embed + positional encoding.
  f32_to_bf16<<<4096, 256, 0, stream>>>(dw, dw16, V_ * D_);
  embed_pe<<<2048, 256, 0, stream>>>(src, pe, emb, xf, xb16);

  for (int l = 0; l < L_; ++l) {
    const size_t wo = (size_t)l * D_ * D_;
    f32_to_bf16<<<1024, 256, 0, stream>>>(qw + wo, wq16, D_ * D_);
    f32_to_bf16<<<1024, 256, 0, stream>>>(kw + wo, wk16, D_ * D_);
    f32_to_bf16<<<1024, 256, 0, stream>>>(vw + wo, wv16, D_ * D_);

    // Q = x @ qw^T + qb   [2048,1024]  (bias per column)
    gemm_bt<1, true, false><<<dim3(D_ / BN, NTOK / BM, 1), 256, 0, stream>>>(
        xb16, 0, D_, wq16, 0, D_, q16, 0, D_, qb + (size_t)l * D_, D_);
    // K = x @ kw^T + kb
    gemm_bt<1, true, false><<<dim3(D_ / BN, NTOK / BM, 1), 256, 0, stream>>>(
        xb16, 0, D_, wk16, 0, D_, k16, 0, D_, kb + (size_t)l * D_, D_);
    // V^T = vw @ x_b^T + vb  -> vt[b][d][t]  (bias per row = output feature)
    gemm_bt<2, true, false><<<dim3(T_ / BN, D_ / BM, B_), 256, 0, stream>>>(
        wv16, 0, D_, xb16, TD, D_, vt16, DT, T_, vb + (size_t)l * D_, D_);
    // scores = Q K^T (unscaled), skip fully-masked tiles
    gemm_bt<0, false, true><<<dim3(T_ / BN, T_ / BM, B_), 256, 0, stream>>>(
        q16, TD, D_, k16, TD, D_, sc, TT, T_, nullptr, D_);
    // causal softmax -> bf16 probabilities (masked entries = 0)
    softmax_causal<<<dim3(T_, B_), 256, 0, stream>>>(sc, pr16);
    // o = P @ V  (= P @ (V^T)^T, both operands K-contiguous)
    gemm_bt<0, false, false><<<dim3(D_ / BN, T_ / BM, B_), 256, 0, stream>>>(
        pr16, TT, T_, vt16, DT, T_, of, TD, D_, nullptr, T_);
    // x += LN(o) * lnw + lnb
    ln_residual<<<NTOK, 256, 0, stream>>>(xf, xb16, of,
                                          lnw + (size_t)l * D_, lnb + (size_t)l * D_);
  }

  // final LayerNorm then logits = xn @ dw^T + db  [2048, 32000]
  final_ln<<<NTOK, 256, 0, stream>>>(xf, xnb16, fw, fb);
  gemm_bt<1, false, false><<<dim3(V_ / BN, NTOK / BM, 1), 256, 0, stream>>>(
      xnb16, 0, D_, dw16, 0, D_, out, 0, V_, db, D_);
}
